// GPT_68496138436734
// MI455X (gfx1250) — matrix operand order repaired and checked by reading, not compiled, not run on hardware
//
#include <hip/hip_runtime.h>
#include <hip/hip_bf16.h>
#include <math.h>
#include <stdint.h>

// ---------------------------------------------------------------------------
// GPT-2 small forward (B=8,T=1024,D=768,H=12,L=12) for gfx1250 (MI455X).
// bf16 WMMA (v_wmma_f32_16x16x32_bf16) for all GEMMs + flash attention.
// A and B tiles staged with gfx1250 async global->LDS DMA (ASYNCcnt).
// Weights pre-converted once per launch to transposed bf16 Wt[N][K].
// ---------------------------------------------------------------------------

typedef __attribute__((ext_vector_type(16))) __bf16          v16bf;
typedef __attribute__((ext_vector_type(8)))  float           v8f;
typedef __attribute__((ext_vector_type(8)))  unsigned short  us8;

union FragB { v16bf v; unsigned short s[16]; };

__device__ __forceinline__ unsigned short f2bf(float f) {
    unsigned u = __float_as_uint(f);
    u += 0x7fffu + ((u >> 16) & 1u);          // round-to-nearest-even
    return (unsigned short)(u >> 16);
}
__device__ __forceinline__ float bf2f(unsigned short s) {
    return __uint_as_float(((unsigned)s) << 16);
}
__device__ __forceinline__ v8f wmma_bf16(v16bf a, v16bf b, v8f c) {
    return __builtin_amdgcn_wmma_f32_16x16x32_bf16(
        false, a, false, b, (short)0, c, false, false);
}

// ---------------------------------------------------------------------------
// Weight convert+transpose: W[K][N] f32 -> Wt[N][K] bf16 (once per launch)
// ---------------------------------------------------------------------------
__global__ __launch_bounds__(256) void wconv_kernel(
    const float* __restrict__ W, unsigned short* __restrict__ Wt, int K, int N)
{
    __shared__ unsigned short t[64 * 66];
    const size_t mat = (size_t)blockIdx.z * K * N;
    const int k0 = blockIdx.y * 64, n0 = blockIdx.x * 64;
    const int tid = threadIdx.x;
    const int r = tid >> 2, s = tid & 3;
    const float* wp = W + mat + (size_t)(k0 + r) * N + n0 + s * 16;
#pragma unroll
    for (int i = 0; i < 16; ++i) t[r * 66 + s * 16 + i] = f2bf(wp[i]);
    __syncthreads();
    unsigned short* op = Wt + mat + (size_t)(n0 + r) * K + k0 + s * 16;
#pragma unroll
    for (int i = 0; i < 16; ++i) op[i] = t[(s * 16 + i) * 66 + r];
}

// ---------------------------------------------------------------------------
// Embedding: h[b*T+t, :] = wte[x[b,t], :] + wpe[t, :]
// ---------------------------------------------------------------------------
__global__ __launch_bounds__(256) void embed_kernel(
    const int* __restrict__ x, const float* __restrict__ wte,
    const float* __restrict__ wpe, float* __restrict__ h, int T)
{
    const int row = blockIdx.x;
    const int t   = row % T;
    const int tok = x[row];
    const int tid = threadIdx.x;
#pragma unroll
    for (int i = 0; i < 3; ++i) {
        int d = tid + i * 256;
        h[(size_t)row * 768 + d] = wte[(size_t)tok * 768 + d] + wpe[(size_t)t * 768 + d];
    }
}

// ---------------------------------------------------------------------------
// LayerNorm row kernel (D=768) -> bf16 output for WMMA A operand
// ---------------------------------------------------------------------------
__global__ __launch_bounds__(256) void layernorm_kernel(
    const float* __restrict__ x, const float* __restrict__ gam,
    const float* __restrict__ bet, unsigned short* __restrict__ out)
{
    __shared__ float red[256];
    const int row = blockIdx.x, tid = threadIdx.x;
    const float* xr = x + (size_t)row * 768;
    float v[3];
#pragma unroll
    for (int i = 0; i < 3; ++i) v[i] = xr[tid + i * 256];
    red[tid] = v[0] + v[1] + v[2];
    __syncthreads();
    for (int off = 128; off > 0; off >>= 1) {
        if (tid < off) red[tid] += red[tid + off];
        __syncthreads();
    }
    const float mu = red[0] * (1.0f / 768.0f);
    __syncthreads();
    float sq = 0.f;
#pragma unroll
    for (int i = 0; i < 3; ++i) { float c = v[i] - mu; sq += c * c; }
    red[tid] = sq;
    __syncthreads();
    for (int off = 128; off > 0; off >>= 1) {
        if (tid < off) red[tid] += red[tid + off];
        __syncthreads();
    }
    const float inv = rsqrtf(red[0] * (1.0f / 768.0f) + 1e-5f);
#pragma unroll
    for (int i = 0; i < 3; ++i) {
        int d = tid + i * 256;
        out[(size_t)row * 768 + d] = f2bf((v[i] - mu) * inv * gam[d] + bet[d]);
    }
}

// ---------------------------------------------------------------------------
// Tiled WMMA GEMM: C[M,N] = A_bf16[M,K] * Wt_bf16[N,K]^T + bias (+res)(+gelu)
// Block tile 128x256, 8 waves, each wave 64x64 = 4x4 WMMA tiles.
// Double-buffered LDS, one barrier per K-step, both tiles via async DMA.
// ---------------------------------------------------------------------------
template <bool RES, bool OUTF, bool OUTB, bool GELU>
__global__ __launch_bounds__(256) void gemm_kernel(
    const unsigned short* __restrict__ A, const unsigned short* __restrict__ Wt,
    const float* __restrict__ bias, const float* __restrict__ residual,
    float* __restrict__ outF, unsigned short* __restrict__ outB,
    int N, int K)
{
    __shared__ unsigned short As[2][128 * 40];   // [m][k], pitch 40 (80B)
    __shared__ unsigned short Bs[2][256 * 40];   // [n][k], pitch 40

    const int m0 = blockIdx.y * 128;
    const int n0 = blockIdx.x * 256;
    const int tid  = threadIdx.x;
    const int wave = tid >> 5, lane = tid & 31;
    const int wm = wave & 1, wn = wave >> 1;     // 2 (M) x 4 (N) waves
    const int lhalf = lane >> 4, l15 = lane & 15;

    const int arow = tid >> 1, ahalf = tid & 1;
    const unsigned long long abase = (unsigned long long)(uintptr_t)A;
    const unsigned long long bbase = (unsigned long long)(uintptr_t)Wt;

    v8f zero = {};
    v8f acc[4][4];
#pragma unroll
    for (int f = 0; f < 4; ++f)
#pragma unroll
        for (int bf = 0; bf < 4; ++bf) acc[f][bf] = zero;

    // issue async DMA for K-step `it` (A: 2x16B, B: 4x16B per thread)
    auto issue_stage = [&](int it) {
        const int kt  = it << 5;
        const int buf = it & 1;
        unsigned lds_a = (unsigned)(uintptr_t)&As[buf][arow * 40 + ahalf * 16];
        unsigned voffa = (unsigned)(((m0 + arow) * K + kt + ahalf * 16) * 2);
        asm volatile("global_load_async_to_lds_b128 %0, %1, %2"
                     :: "v"(lds_a), "v"(voffa), "s"(abase) : "memory");
        asm volatile("global_load_async_to_lds_b128 %0, %1, %2 offset:16"
                     :: "v"(lds_a), "v"(voffa), "s"(abase) : "memory");
        unsigned lds_b = (unsigned)(uintptr_t)&Bs[buf][tid * 40];
        unsigned voffb = (unsigned)(((n0 + tid) * K + kt) * 2);
        asm volatile("global_load_async_to_lds_b128 %0, %1, %2"
                     :: "v"(lds_b), "v"(voffb), "s"(bbase) : "memory");
        asm volatile("global_load_async_to_lds_b128 %0, %1, %2 offset:16"
                     :: "v"(lds_b), "v"(voffb), "s"(bbase) : "memory");
        asm volatile("global_load_async_to_lds_b128 %0, %1, %2 offset:32"
                     :: "v"(lds_b), "v"(voffb), "s"(bbase) : "memory");
        asm volatile("global_load_async_to_lds_b128 %0, %1, %2 offset:48"
                     :: "v"(lds_b), "v"(voffb), "s"(bbase) : "memory");
    };

    const int niter = K >> 5;
    issue_stage(0);
    asm volatile("s_wait_asynccnt 0x0" ::: "memory");
    __syncthreads();

    for (int it = 0; it < niter; ++it) {
        const int buf = it & 1;
        if (it + 1 < niter) issue_stage(it + 1);

        FragB af[4], bfr[4];
#pragma unroll
        for (int f = 0; f < 4; ++f) {
            const unsigned short* p = &As[buf][(wm * 64 + f * 16 + l15) * 40];
            int koff = lhalf * 16;       // A layout: K0-15/K16-31
#pragma unroll
            for (int i = 0; i < 8; ++i) af[f].s[i]     = p[koff + i];
#pragma unroll
            for (int i = 0; i < 8; ++i) af[f].s[8 + i] = p[koff + 8 + i];
        }
#pragma unroll
        for (int bf = 0; bf < 4; ++bf) {
            const unsigned short* p =
                &Bs[buf][(wn * 64 + bf * 16 + l15) * 40 + lhalf * 16];
#pragma unroll
            for (int i = 0; i < 16; ++i) bfr[bf].s[i] = p[i];  // B: K0-15/K16-31
        }
#pragma unroll
        for (int f = 0; f < 4; ++f)
#pragma unroll
            for (int bf = 0; bf < 4; ++bf)
                acc[f][bf] = wmma_bf16(af[f].v, bfr[bf].v, acc[f][bf]);

        asm volatile("s_wait_asynccnt 0x0" ::: "memory");
        __syncthreads();
    }

    // ---- epilogue: bias / residual / gelu / stores (fully specialized) ----
    float bias_v[4];
#pragma unroll
    for (int bf = 0; bf < 4; ++bf) bias_v[bf] = bias[n0 + wn * 64 + bf * 16 + l15];
#pragma unroll
    for (int f = 0; f < 4; ++f) {
#pragma unroll
        for (int g = 0; g < 8; ++g) {
            int m = m0 + wm * 64 + f * 16 + g + lhalf * 8;
#pragma unroll
            for (int bf = 0; bf < 4; ++bf) {
                int n = n0 + wn * 64 + bf * 16 + l15;
                float v = acc[f][bf][g] + bias_v[bf];
                if constexpr (RES)  v += residual[(size_t)m * N + n];
                if constexpr (GELU) v = 0.5f * v * (1.0f + erff(v * 0.70710678118654752f));
                if constexpr (OUTF) outF[(size_t)m * N + n] = v;
                if constexpr (OUTB) outB[(size_t)m * N + n] = f2bf(v);
            }
        }
    }
}

// ---------------------------------------------------------------------------
// Flash attention, causal, NO 1/sqrt(hd) scaling (faithful to reference).
// 128 q rows per block (16 per wave), kv chunks of 64: 16 WMMAs per chunk.
// ---------------------------------------------------------------------------
__global__ __launch_bounds__(256) void flash_kernel(
    const unsigned short* __restrict__ qkv, unsigned short* __restrict__ o,
    int T, int Hn)
{
    __shared__ unsigned short Ks[64 * 80];        // [kv][hd], pitch 80
    __shared__ unsigned short VT[64 * 80];        // [hd][kv], pitch 80
    __shared__ unsigned short Ps[8 * 16 * 68];    // per-wave P scratch [q][kv]

    const int bh = blockIdx.y;
    const int b  = bh / Hn, h = bh % Hn;
    const int q0 = blockIdx.x * 128;
    const int tid = threadIdx.x, wave = tid >> 5, lane = tid & 31;
    const int lhalf = lane >> 4, l15 = lane & 15;
    const int qrow  = q0 + wave * 16;

    // ---- load Q as two A-frags (head_dim 64 -> 2 K-steps of 32) ----
    FragB aq[2];
    {
        const unsigned short* qp =
            qkv + (size_t)(b * T + qrow + l15) * 2304 + h * 64;
#pragma unroll
        for (int ks = 0; ks < 2; ++ks) {
            int koff = ks * 32 + lhalf * 16;
#pragma unroll
            for (int i = 0; i < 8; ++i) aq[ks].s[i]     = qp[koff + i];
#pragma unroll
            for (int i = 0; i < 8; ++i) aq[ks].s[8 + i] = qp[koff + 8 + i];
        }
    }

    float m_i[8], l_i[8];
#pragma unroll
    for (int g = 0; g < 8; ++g) { m_i[g] = -3.0e38f; l_i[g] = 0.0f; }
    v8f zero = {};
    v8f accO[4];
#pragma unroll
    for (int t4 = 0; t4 < 4; ++t4) accO[t4] = zero;

    const int nch   = (q0 + 128) / 64;            // causal bound for block
    const int kvr   = tid >> 2, seg = tid & 3;    // staging map: 16 ush/thread

    for (int ci = 0; ci < nch; ++ci) {
        const int kv0 = ci * 64;
        // ---- stage K [kv][hd] and V transposed [hd][kv] ----
        {
            const size_t base = (size_t)(b * T + kv0 + kvr) * 2304;
            const unsigned short* kp = qkv + base + 768 + h * 64 + seg * 16;
            us8* kd = (us8*)&Ks[kvr * 80 + seg * 16];
            kd[0] = ((const us8*)kp)[0];
            kd[1] = ((const us8*)kp)[1];
            const unsigned short* vp = qkv + base + 1536 + h * 64 + seg * 16;
#pragma unroll
            for (int i = 0; i < 16; ++i) VT[(seg * 16 + i) * 80 + kvr] = vp[i];
        }
        __syncthreads();

        if (kv0 <= qrow + 15) {                   // wave-uniform
            // ---- S[16q x 64kv] = Q * K^T : 8 WMMAs ----
            v8f s[4];
#pragma unroll
            for (int nt = 0; nt < 4; ++nt) {
                s[nt] = zero;
#pragma unroll
                for (int ks = 0; ks < 2; ++ks) {
                    FragB bk;
                    const unsigned short* p =
                        &Ks[(nt * 16 + l15) * 80 + ks * 32 + lhalf * 16];
#pragma unroll
                    for (int i = 0; i < 16; ++i) bk.s[i] = p[i];
                    s[nt] = wmma_bf16(aq[ks].v, bk.v, s[nt]);
                }
            }
            // ---- causal mask ----
#pragma unroll
            for (int nt = 0; nt < 4; ++nt) {
                int kvc = kv0 + nt * 16 + l15;
#pragma unroll
                for (int g = 0; g < 8; ++g) {
                    int qr = qrow + g + lhalf * 8;
                    s[nt][g] = (kvc > qr) ? -3.0e38f : s[nt][g];
                }
            }
            // ---- online softmax ----
#pragma unroll
            for (int g = 0; g < 8; ++g) {
                float mx = fmaxf(fmaxf(s[0][g], s[1][g]), fmaxf(s[2][g], s[3][g]));
#pragma unroll
                for (int d = 1; d < 16; d <<= 1) mx = fmaxf(mx, __shfl_xor(mx, d, 32));
                float mn = fmaxf(m_i[g], mx);
                float sc = __expf(m_i[g] - mn);
                float p0 = __expf(s[0][g] - mn);
                float p1 = __expf(s[1][g] - mn);
                float p2 = __expf(s[2][g] - mn);
                float p3 = __expf(s[3][g] - mn);
                float sum = (p0 + p1) + (p2 + p3);
#pragma unroll
                for (int d = 1; d < 16; d <<= 1) sum += __shfl_xor(sum, d, 32);
                l_i[g] = l_i[g] * sc + sum;
                m_i[g] = mn;
                s[0][g] = p0; s[1][g] = p1; s[2][g] = p2; s[3][g] = p3;
#pragma unroll
                for (int t4 = 0; t4 < 4; ++t4) accO[t4][g] *= sc;
            }
            // ---- relayout P: C-layout -> LDS -> A-layout (same wave) ----
            unsigned short* pw = &Ps[wave * 16 * 68];
#pragma unroll
            for (int nt = 0; nt < 4; ++nt)
#pragma unroll
                for (int g = 0; g < 8; ++g)
                    pw[(g + lhalf * 8) * 68 + nt * 16 + l15] = f2bf(s[nt][g]);
            FragB pa[2];
#pragma unroll
            for (int kp2 = 0; kp2 < 2; ++kp2) {
                const unsigned short* p = &pw[l15 * 68 + kp2 * 32 + lhalf * 16];
#pragma unroll
                for (int i = 0; i < 8; ++i) pa[kp2].s[i]     = p[i];
#pragma unroll
                for (int i = 0; i < 8; ++i) pa[kp2].s[8 + i] = p[8 + i];
            }
            // ---- O[16 x 64] += P[16x64] * V[64x64] : 8 WMMAs ----
#pragma unroll
            for (int t4 = 0; t4 < 4; ++t4) {
#pragma unroll
                for (int kp2 = 0; kp2 < 2; ++kp2) {
                    FragB bv;
                    const unsigned short* p =
                        &VT[(t4 * 16 + l15) * 80 + kp2 * 32 + lhalf * 16];
#pragma unroll
                    for (int i = 0; i < 16; ++i) bv.s[i] = p[i];
                    accO[t4] = wmma_bf16(pa[kp2].v, bv.v, accO[t4]);
                }
            }
        }
        __syncthreads();
    }

    // ---- normalize and store O as bf16 ----
#pragma unroll
    for (int t4 = 0; t4 < 4; ++t4) {
#pragma unroll
        for (int g = 0; g < 8; ++g) {
            int qr = qrow + g + lhalf * 8;
            int hd = t4 * 16 + l15;
            float v = accO[t4][g] / l_i[g];
            o[(size_t)(b * T + qr) * 768 + h * 64 + hd] = f2bf(v);
        }
    }
}

// ---------------------------------------------------------------------------
// Tied LM head, last position only: out[b,v] = dot(hn[b*T+T-1,:], wte[v,:])
// ---------------------------------------------------------------------------
__global__ __launch_bounds__(256) void lmhead_kernel(
    const unsigned short* __restrict__ hn, const float* __restrict__ wte,
    float* __restrict__ out, int T, int V)
{
    __shared__ float red[256];
    const int v = blockIdx.x, tid = threadIdx.x;
    float acc[8];
#pragma unroll
    for (int b = 0; b < 8; ++b) acc[b] = 0.f;
    for (int d = tid; d < 768; d += 256) {
        float w = wte[(size_t)v * 768 + d];
#pragma unroll
        for (int b = 0; b < 8; ++b)
            acc[b] += w * bf2f(hn[(size_t)(b * T + T - 1) * 768 + d]);
    }
#pragma unroll
    for (int b = 0; b < 8; ++b) {
        red[tid] = acc[b];
        __syncthreads();
        for (int off = 128; off > 0; off >>= 1) {
            if (tid < off) red[tid] += red[tid + off];
            __syncthreads();
        }
        if (tid == 0) out[(size_t)b * V + v] = red[0];
        __syncthreads();
    }
}

// ---------------------------------------------------------------------------
// Host orchestration
// ---------------------------------------------------------------------------
extern "C" void kernel_launch(void* const* d_in, const int* in_sizes, int n_in,
                              void* d_out, int out_size, void* d_ws, size_t ws_size,
                              hipStream_t stream)
{
    (void)in_sizes; (void)n_in; (void)out_size; (void)ws_size;
    constexpr int B = 8, T = 1024, D = 768, H = 12, L = 12, V = 50257;
    constexpr int M = B * T;

    const int*   x       = (const int*)  d_in[0];
    const float* wte     = (const float*)d_in[1];
    const float* wpe     = (const float*)d_in[2];
    const float* ln1_g   = (const float*)d_in[3];
    const float* ln1_b   = (const float*)d_in[4];
    const float* w_attn  = (const float*)d_in[5];
    const float* b_attn  = (const float*)d_in[6];
    const float* w_aproj = (const float*)d_in[7];
    const float* b_aproj = (const float*)d_in[8];
    const float* ln2_g   = (const float*)d_in[9];
    const float* ln2_b   = (const float*)d_in[10];
    const float* w_fc    = (const float*)d_in[11];
    const float* b_fc    = (const float*)d_in[12];
    const float* w_mproj = (const float*)d_in[13];
    const float* b_mproj = (const float*)d_in[14];
    const float* lnf_g   = (const float*)d_in[15];
    const float* lnf_b   = (const float*)d_in[16];

    char* ws = (char*)d_ws;
    size_t off = 0;
    float*          h    = (float*)(ws + off);          off += (size_t)M * D * 4;
    unsigned short* hnb  = (unsigned short*)(ws + off); off += (size_t)M * D * 2;
    unsigned short* qkvb = (unsigned short*)(ws + off); off += (size_t)M * 3 * D * 2;
    unsigned short* ob   = (unsigned short*)(ws + off); off += (size_t)M * D * 2;
    unsigned short* fcb  = (unsigned short*)(ws + off); off += (size_t)M * 4 * D * 2;
    unsigned short* wt_attn  = (unsigned short*)(ws + off); off += (size_t)L * D * 3 * D * 2;
    unsigned short* wt_aproj = (unsigned short*)(ws + off); off += (size_t)L * D * D * 2;
    unsigned short* wt_fc    = (unsigned short*)(ws + off); off += (size_t)L * D * 4 * D * 2;
    unsigned short* wt_mproj = (unsigned short*)(ws + off); off += (size_t)L * 4 * D * D * 2;

    // pre-convert all weights to transposed bf16 (deterministic, every call)
    wconv_kernel<<<dim3(3 * D / 64, D / 64, L), 256, 0, stream>>>(w_attn,  wt_attn,  D, 3 * D);
    wconv_kernel<<<dim3(D / 64, D / 64, L),     256, 0, stream>>>(w_aproj, wt_aproj, D, D);
    wconv_kernel<<<dim3(4 * D / 64, D / 64, L), 256, 0, stream>>>(w_fc,    wt_fc,    D, 4 * D);
    wconv_kernel<<<dim3(D / 64, 4 * D / 64, L), 256, 0, stream>>>(w_mproj, wt_mproj, 4 * D, D);

    embed_kernel<<<M, 256, 0, stream>>>(x, wte, wpe, h, T);

    for (int l = 0; l < L; ++l) {
        layernorm_kernel<<<M, 256, 0, stream>>>(h, ln1_g + l * D, ln1_b + l * D, hnb);
        gemm_kernel<false, false, true, false>
            <<<dim3(3 * D / 256, M / 128), 256, 0, stream>>>(
            hnb, wt_attn + (size_t)l * D * 3 * D, b_attn + (size_t)l * 3 * D,
            nullptr, nullptr, qkvb, 3 * D, D);
        flash_kernel<<<dim3(T / 128, B * H), 256, 0, stream>>>(qkvb, ob, T, H);
        gemm_kernel<true, true, false, false>
            <<<dim3(D / 256, M / 128), 256, 0, stream>>>(
            ob, wt_aproj + (size_t)l * D * D, b_aproj + (size_t)l * D,
            h, h, nullptr, D, D);
        layernorm_kernel<<<M, 256, 0, stream>>>(h, ln2_g + l * D, ln2_b + l * D, hnb);
        gemm_kernel<false, false, true, true>
            <<<dim3(4 * D / 256, M / 128), 256, 0, stream>>>(
            hnb, wt_fc + (size_t)l * D * 4 * D, b_fc + (size_t)l * 4 * D,
            nullptr, nullptr, fcb, 4 * D, D);
        gemm_kernel<true, true, false, false>
            <<<dim3(D / 256, M / 128), 256, 0, stream>>>(
            fcb, wt_mproj + (size_t)l * 4 * D * D, b_mproj + (size_t)l * D,
            h, h, nullptr, D, 4 * D);
    }

    layernorm_kernel<<<M, 256, 0, stream>>>(h, lnf_g, lnf_b, hnb);
    lmhead_kernel<<<V, 256, 0, stream>>>(hnb, wte, (float*)d_out, T, V);
}
